// PCDetDynamicVoxelVFE_60000693125783
// MI455X (gfx1250) — matrix operand order repaired and checked by reading, not compiled, not run on hardware
//
#include <hip/hip_runtime.h>
#include <hip/hip_bf16.h>
#include <stdint.h>

typedef __attribute__((ext_vector_type(16))) _Float16 v16h;
typedef __attribute__((ext_vector_type(8)))  _Float16 v8h;
typedef __attribute__((ext_vector_type(8)))  float    v8f;
typedef __attribute__((ext_vector_type(4)))  unsigned v4u;
typedef __attribute__((ext_vector_type(8)))  int      v8i;
typedef __attribute__((ext_vector_type(4)))  int      v4i;

#define NPTS      400000
#define GX        704
#define GY        800
#define GZ        20
#define SCALE_XYZ (GX*GY*GZ)      /* 11,264,000 */
#define SCALE_YZ  (GY*GZ)         /* 16,000     */
#define SCALE_Z   (GZ)
#define NKEYS     (2*SCALE_XYZ)   /* 22,528,000 */
#define SCAN_TILE 4096
#define NTILES    (NKEYS/SCAN_TILE) /* 5500 */
#define BN_EPS    0.001f

static_assert(NPTS % 64 == 0, "tiling assumes N % 64 == 0");
static_assert(NKEYS % SCAN_TILE == 0, "scan assumes exact tiles");

// ------------------------------------------------ TDM: global -> LDS DMA ----
// Stages `dwords` DWORDs from gptr to LDS offset 0 via the Tensor Data Mover.
// 2-D descriptor: tile_dim0 = tensor_dim0 = dwords (elements of 4B),
// tile_dim1 = tensor_dim1 = 1, groups 2/3 (and the trailing group of the
// 6-arg clang-23 builtin) zero => tile_dim3/4 unused, 2-D semantics.
// Wave 0 issues (wave-uniform branch skips issue on waves 1..3), waits
// TENSORcnt, then the workgroup barrier publishes the LDS contents.
__device__ __forceinline__ void tdm_load_lds(const void* gptr, unsigned dwords) {
  if ((threadIdx.x >> 5) == 0) {
    unsigned long long ga = (unsigned long long)gptr;
    v4u g0;
    g0[0] = 1u;                                        // count=1, user mode
    g0[1] = 0u;                                        // lds_addr = 0 (sB at base)
    g0[2] = (unsigned)(ga & 0xFFFFFFFFu);              // global_addr[31:0]
    g0[3] = (unsigned)((ga >> 32) & 0x01FFFFFFu)       // global_addr[56:32]
          | 0x80000000u;                               // type=2 ("image")
    v8i g1;
    g1[0] = (int)(2u << 16);                           // data_size=2 (4 bytes)
    g1[1] = (int)((dwords & 0xFFFFu) << 16);           // tensor_dim0[15:0]
    g1[2] = (int)(((dwords >> 16) & 0xFFFFu)           // tensor_dim0[31:16]
          | (1u << 16));                               // tensor_dim1[15:0]=1
    g1[3] = (int)((dwords & 0xFFFFu) << 16);           // tile_dim0
    g1[4] = 1;                                         // tile_dim1=1, tile_dim2=0
    g1[5] = (int)dwords;                               // tensor_dim0_stride lo
    g1[6] = 0;
    g1[7] = 0;
    v4i g2 = {0, 0, 0, 0};
    v4i g3 = {0, 0, 0, 0};
    v8i g4 = {0, 0, 0, 0, 0, 0, 0, 0};
    __builtin_amdgcn_tensor_load_to_lds(g0, g1, g2, g3, g4, 0);
    __builtin_amdgcn_s_wait_tensorcnt(0);
  }
  __syncthreads();
}

// ---------------------------------------------------------------- fill ------
__global__ void k_fill_u4(uint4* __restrict__ p, unsigned v, long n4) {
  long i = (long)blockIdx.x * blockDim.x + threadIdx.x;
  long stride = (long)gridDim.x * blockDim.x;
  uint4 q = make_uint4(v, v, v, v);
  for (; i < n4; i += stride) p[i] = q;
}

// ---------------------------------------------------------------- keys ------
__global__ void k_keys(const float* __restrict__ pts, int* __restrict__ key,
                       unsigned* __restrict__ flags) {
  int i = blockIdx.x * blockDim.x + threadIdx.x;
  if (i >= NPTS) return;
  float b = pts[i*5+0], x = pts[i*5+1], y = pts[i*5+2], z = pts[i*5+3];
  int cx = (int)floorf((x - 0.0f)   / 0.1f);
  int cy = (int)floorf((y - -40.0f) / 0.1f);
  int cz = (int)floorf((z - -3.0f)  / 0.2f);
  int k = (int)b * SCALE_XYZ + cx * SCALE_YZ + cy * SCALE_Z + cz;
  key[i] = k;
  flags[k] = 1u;
}

// ------------------------------------------------------------ 3-pass scan ---
__global__ void k_scanA(unsigned* __restrict__ data, unsigned* __restrict__ tilesums) {
  __shared__ unsigned sh[256];
  int t = threadIdx.x;
  size_t base = (size_t)blockIdx.x * SCAN_TILE + (size_t)t * 16;
  uint4* p = (uint4*)(data + base);
  uint4 a0 = p[0], a1 = p[1], a2 = p[2], a3 = p[3];
  unsigned vals[16] = {a0.x,a0.y,a0.z,a0.w, a1.x,a1.y,a1.z,a1.w,
                       a2.x,a2.y,a2.z,a2.w, a3.x,a3.y,a3.z,a3.w};
  unsigned ex[16], run = 0;
  #pragma unroll
  for (int j = 0; j < 16; ++j) { ex[j] = run; run += vals[j]; }
  sh[t] = run;
  __syncthreads();
  for (int off = 1; off < 256; off <<= 1) {
    unsigned v = (t >= off) ? sh[t-off] : 0u;
    __syncthreads();
    sh[t] += v;
    __syncthreads();
  }
  unsigned texcl = sh[t] - run;
  uint4 o0 = make_uint4(texcl+ex[0], texcl+ex[1], texcl+ex[2], texcl+ex[3]);
  uint4 o1 = make_uint4(texcl+ex[4], texcl+ex[5], texcl+ex[6], texcl+ex[7]);
  uint4 o2 = make_uint4(texcl+ex[8], texcl+ex[9], texcl+ex[10], texcl+ex[11]);
  uint4 o3 = make_uint4(texcl+ex[12], texcl+ex[13], texcl+ex[14], texcl+ex[15]);
  p[0] = o0; p[1] = o1; p[2] = o2; p[3] = o3;
  if (t == 255) tilesums[blockIdx.x] = sh[255];
}

__global__ void k_scanB(unsigned* __restrict__ tilesums, int n) {
  __shared__ unsigned sh[256];
  __shared__ unsigned srun;
  int t = threadIdx.x;
  if (t == 0) srun = 0;
  __syncthreads();
  for (int base = 0; base < n; base += 256) {
    int i = base + t;
    unsigned v = (i < n) ? tilesums[i] : 0u;
    sh[t] = v;
    __syncthreads();
    for (int off = 1; off < 256; off <<= 1) {
      unsigned u = (t >= off) ? sh[t-off] : 0u;
      __syncthreads();
      sh[t] += u;
      __syncthreads();
    }
    unsigned excl = sh[t] - v;
    unsigned offs = srun;
    __syncthreads();
    if (i < n) tilesums[i] = excl + offs;
    if (t == 255) srun = offs + sh[255];
    __syncthreads();
  }
}

__global__ void k_scanC(unsigned* __restrict__ data, const unsigned* __restrict__ tilesums) {
  size_t base = (size_t)blockIdx.x * SCAN_TILE + (size_t)threadIdx.x * 16;
  unsigned off = tilesums[blockIdx.x];
  uint4* p = (uint4*)(data + base);
  #pragma unroll
  for (int k = 0; k < 4; ++k) {
    uint4 a = p[k];
    a.x += off; a.y += off; a.z += off; a.w += off;
    p[k] = a;
  }
}

// ---------------------------------------------------- inverse + seg sums ----
__global__ void k_inv(const int* __restrict__ key, const unsigned* __restrict__ rank,
                      int* __restrict__ inv, int* __restrict__ counts,
                      float* __restrict__ sums, int* __restrict__ unq,
                      const float* __restrict__ pts) {
  int i = blockIdx.x * blockDim.x + threadIdx.x;
  if (i >= NPTS) return;
  int k = key[i];
  int v = (int)rank[k];          // # distinct keys < k  == sorted-unique inverse
  inv[i] = v;
  unq[v] = k;                    // idempotent scatter
  atomicAdd(counts + v, 1);
  atomicAdd(sums + (size_t)v*3 + 0, pts[i*5+1]);
  atomicAdd(sums + (size_t)v*3 + 1, pts[i*5+2]);
  atomicAdd(sums + (size_t)v*3 + 2, pts[i*5+3]);
}

// -------------------------------------------------- 10-feature build (f16) --
__global__ void k_feats(const float* __restrict__ pts, const int* __restrict__ inv,
                        const int* __restrict__ counts, const float* __restrict__ sums,
                        _Float16* __restrict__ feats) {
  int i = blockIdx.x * blockDim.x + threadIdx.x;
  if (i >= NPTS) return;
  float x = pts[i*5+1], y = pts[i*5+2], z = pts[i*5+3], inten = pts[i*5+4];
  int v = inv[i];
  float cnt = fmaxf((float)counts[v], 1.0f);
  float mx = sums[(size_t)v*3+0] / cnt;
  float my = sums[(size_t)v*3+1] / cnt;
  float mz = sums[(size_t)v*3+2] / cnt;
  int cx = (int)floorf((x - 0.0f)   / 0.1f);
  int cy = (int)floorf((y - -40.0f) / 0.1f);
  int cz = (int)floorf((z - -3.0f)  / 0.2f);
  float fx = x - ((float)cx * 0.1f + 0.05f + 0.0f);
  float fy = y - ((float)cy * 0.1f + 0.05f + -40.0f);
  float fz = z - ((float)cz * 0.2f + 0.10f + -3.0f);
  union { _Float16 h[32]; uint4 u[4]; } buf;
  #pragma unroll
  for (int j = 0; j < 32; ++j) buf.h[j] = (_Float16)0.0f;
  buf.h[0] = (_Float16)x;  buf.h[1] = (_Float16)y;  buf.h[2] = (_Float16)z;
  buf.h[3] = (_Float16)inten;
  buf.h[4] = (_Float16)(x - mx); buf.h[5] = (_Float16)(y - my); buf.h[6] = (_Float16)(z - mz);
  buf.h[7] = (_Float16)fx; buf.h[8] = (_Float16)fy; buf.h[9] = (_Float16)fz;
  uint4* dst = (uint4*)(feats + (size_t)i * 32);
  dst[0] = buf.u[0]; dst[1] = buf.u[1]; dst[2] = buf.u[2]; dst[3] = buf.u[3];
}

// ---------------------------------------------- weight -> B-fragment prep ---
// B fragment layout (16-bit B, 32xK16 tile): lane l holds column n = (l&15),
// halves j=0..15 hold k = (l>>4)*8 + (j&7) + (j>>3)*16 (matches A table order).
__global__ void k_prepw0(const float* __restrict__ W0, _Float16* __restrict__ out) {
  int idx = blockIdx.x * blockDim.x + threadIdx.x;   // 4 frags * 512
  if (idx >= 4*512) return;
  int t = idx >> 9, r = idx & 511, lane = r >> 4, j = r & 15;
  int k = (lane >> 4) * 8 + (j & 7) + (j >> 3) * 16;
  int n = t * 16 + (lane & 15);
  float v = (k < 10) ? W0[k*64 + n] : 0.0f;          // K padded 10 -> 32
  out[idx] = (_Float16)v;
}

__global__ void k_prepw1(const float* __restrict__ W1, _Float16* __restrict__ out) {
  int idx = blockIdx.x * blockDim.x + threadIdx.x;   // 32 frags * 512
  if (idx >= 32*512) return;
  int f = idx >> 9, r = idx & 511, lane = r >> 4, j = r & 15;
  int s = f >> 3, t = f & 7;
  int k = s * 32 + (lane >> 4) * 8 + (j & 7) + (j >> 3) * 16;
  int n = t * 16 + (lane & 15);
  out[idx] = (_Float16)W1[k*128 + n];
}

// ------------------------------------------------------- GEMM0: 10->64 ------
__global__ __launch_bounds__(128)
void k_gemm0_stats(const _Float16* __restrict__ X, const _Float16* __restrict__ Wp,
                   float* __restrict__ chsum, float* __restrict__ chsq) {
  __shared__ v16h sB[4*32];
  tdm_load_lds(Wp, 4*512/2);
  int lane = threadIdx.x & 31, wid = threadIdx.x >> 5;
  int mt = blockIdx.x * 4 + wid;
  int r = lane & 15, hi = lane >> 4;
  const _Float16* arow = X + ((size_t)(mt*16 + r)) * 32 + hi*8;
  __builtin_prefetch(arow + 4*16*32, 0, 0);
  union { v16h v; v8h h[2]; } A;
  A.h[0] = *(const v8h*)(arow);
  A.h[1] = *(const v8h*)(arow + 16);
  #pragma unroll
  for (int t = 0; t < 4; ++t) {
    v16h b = sB[t*32 + lane];
    v8f c = {};
    c = __builtin_amdgcn_wmma_f32_16x16x32_f16(false, A.v, false, b, (short)0, c, false, false);
    float s = 0.0f, q = 0.0f;
    #pragma unroll
    for (int v = 0; v < 8; ++v) { float e = c[v]; s += e; q += e*e; }
    s += __shfl_xor(s, 16, 32);
    q += __shfl_xor(q, 16, 32);
    if (lane < 16) {
      atomicAdd(&chsum[t*16 + lane], s);
      atomicAdd(&chsq [t*16 + lane], q);
    }
  }
}

__global__ __launch_bounds__(128)
void k_gemm0_apply(const _Float16* __restrict__ X, const _Float16* __restrict__ Wp,
                   const float* __restrict__ scale, const float* __restrict__ shift,
                   const int* __restrict__ inv, _Float16* __restrict__ x0,
                   int* __restrict__ xmax0) {
  __shared__ v16h sB[4*32];
  tdm_load_lds(Wp, 4*512/2);
  int lane = threadIdx.x & 31, wid = threadIdx.x >> 5;
  int mt = blockIdx.x * 4 + wid;
  int r = lane & 15, hi = lane >> 4;
  const _Float16* arow = X + ((size_t)(mt*16 + r)) * 32 + hi*8;
  __builtin_prefetch(arow + 4*16*32, 0, 0);
  union { v16h v; v8h h[2]; } A;
  A.h[0] = *(const v8h*)(arow);
  A.h[1] = *(const v8h*)(arow + 16);
  int invv[8];
  #pragma unroll
  for (int v = 0; v < 8; ++v) invv[v] = inv[mt*16 + hi*8 + v];
  #pragma unroll
  for (int t = 0; t < 4; ++t) {
    v16h b = sB[t*32 + lane];
    v8f c = {};
    c = __builtin_amdgcn_wmma_f32_16x16x32_f16(false, A.v, false, b, (short)0, c, false, false);
    int n = t*16 + (lane & 15);
    float sc = scale[n], sh = shift[n];
    #pragma unroll
    for (int v = 0; v < 8; ++v) {
      int row = mt*16 + hi*8 + v;
      float y = fmaxf(fmaf(c[v], sc, sh), 0.0f);     // BN + ReLU
      x0[(size_t)row*64 + n] = (_Float16)y;
      atomicMax(xmax0 + (size_t)invv[v]*64 + n, __float_as_int(y)); // y>=0: int cmp exact
    }
  }
}

// ------------------------------------------- concat [x0 | xmax0[inv]] f16 ---
__global__ void k_concat(const _Float16* __restrict__ x0, const float* __restrict__ xmax0,
                         const int* __restrict__ inv, _Float16* __restrict__ x1) {
  long idx = (long)blockIdx.x * blockDim.x + threadIdx.x;  // N*16 chunks of 8 halves
  long i = idx >> 4;
  int ch = (int)(idx & 15);
  if (i >= NPTS) return;
  if (ch < 8) {
    const uint4* s = (const uint4*)(x0 + (size_t)i*64 + ch*8);
    *(uint4*)(x1 + (size_t)i*128 + ch*8) = *s;
  } else {
    const float* s = xmax0 + (size_t)inv[i]*64 + (ch-8)*8;
    union { _Float16 h[8]; uint4 u; } buf;
    #pragma unroll
    for (int j = 0; j < 8; ++j) buf.h[j] = (_Float16)s[j];
    *(uint4*)(x1 + (size_t)i*128 + ch*8) = buf.u;
  }
}

// ------------------------------------------------------ GEMM1: 128->128 -----
__global__ __launch_bounds__(128)
void k_gemm1_stats(const _Float16* __restrict__ X, const _Float16* __restrict__ Wp,
                   float* __restrict__ chsum, float* __restrict__ chsq) {
  __shared__ v16h sB[32*32];                         // 32 KB staged B fragments
  tdm_load_lds(Wp, 32*512/2);
  int lane = threadIdx.x & 31, wid = threadIdx.x >> 5;
  int mt = blockIdx.x * 4 + wid;
  int r = lane & 15, hi = lane >> 4;
  const _Float16* arow = X + ((size_t)(mt*16 + r)) * 128 + hi*8;
  __builtin_prefetch(arow + 4*16*128, 0, 0);
  union { v16h v; v8h h[2]; } A[4];
  #pragma unroll
  for (int s = 0; s < 4; ++s) {
    A[s].h[0] = *(const v8h*)(arow + s*32);
    A[s].h[1] = *(const v8h*)(arow + s*32 + 16);
  }
  #pragma unroll
  for (int t = 0; t < 8; ++t) {
    v8f c = {};
    #pragma unroll
    for (int s = 0; s < 4; ++s)
      c = __builtin_amdgcn_wmma_f32_16x16x32_f16(false, A[s].v, false,
                                                 sB[(s*8 + t)*32 + lane],
                                                 (short)0, c, false, false);
    float sm = 0.0f, q = 0.0f;
    #pragma unroll
    for (int v = 0; v < 8; ++v) { float e = c[v]; sm += e; q += e*e; }
    sm += __shfl_xor(sm, 16, 32);
    q  += __shfl_xor(q,  16, 32);
    if (lane < 16) {
      atomicAdd(&chsum[t*16 + lane], sm);
      atomicAdd(&chsq [t*16 + lane], q);
    }
  }
}

__global__ __launch_bounds__(128)
void k_gemm1_apply(const _Float16* __restrict__ X, const _Float16* __restrict__ Wp,
                   const float* __restrict__ scale, const float* __restrict__ shift,
                   const int* __restrict__ inv, float* __restrict__ outF) {
  __shared__ v16h sB[32*32];
  tdm_load_lds(Wp, 32*512/2);
  int lane = threadIdx.x & 31, wid = threadIdx.x >> 5;
  int mt = blockIdx.x * 4 + wid;
  int r = lane & 15, hi = lane >> 4;
  const _Float16* arow = X + ((size_t)(mt*16 + r)) * 128 + hi*8;
  __builtin_prefetch(arow + 4*16*128, 0, 0);
  union { v16h v; v8h h[2]; } A[4];
  #pragma unroll
  for (int s = 0; s < 4; ++s) {
    A[s].h[0] = *(const v8h*)(arow + s*32);
    A[s].h[1] = *(const v8h*)(arow + s*32 + 16);
  }
  int invv[8];
  #pragma unroll
  for (int v = 0; v < 8; ++v) invv[v] = inv[mt*16 + hi*8 + v];
  int* outI = (int*)outF;
  #pragma unroll
  for (int t = 0; t < 8; ++t) {
    v8f c = {};
    #pragma unroll
    for (int s = 0; s < 4; ++s)
      c = __builtin_amdgcn_wmma_f32_16x16x32_f16(false, A[s].v, false,
                                                 sB[(s*8 + t)*32 + lane],
                                                 (short)0, c, false, false);
    int n = t*16 + (lane & 15);
    float sc = scale[n], sh = shift[n];
    #pragma unroll
    for (int v = 0; v < 8; ++v) {
      float y = fmaxf(fmaf(c[v], sc, sh), 0.0f);     // BN + ReLU
      atomicMax(outI + (size_t)invv[v]*128 + n, __float_as_int(y)); // segment_max
    }
  }
}

// ----------------------------------------------------- BN scale/shift -------
__global__ void k_stats(const float* __restrict__ chsum, const float* __restrict__ chsq,
                        const float* __restrict__ gamma, const float* __restrict__ beta,
                        float* __restrict__ scale, float* __restrict__ shift, int C) {
  int c = blockIdx.x * blockDim.x + threadIdx.x;
  if (c >= C) return;
  const float invN = 1.0f / (float)NPTS;
  float mean = chsum[c] * invN;
  float var  = chsq[c] * invN - mean * mean;         // ddof=0, matches jnp.var
  float sc = gamma[c] * rsqrtf(var + BN_EPS);
  scale[c] = sc;
  shift[c] = beta[c] - mean * sc;
}

// ----------------------------------------------------- coord decode ---------
__device__ __forceinline__ int fdivi(int a, int b) {   // floor division (b>0)
  int q = a / b;
  if ((a % b) != 0 && ((a ^ b) < 0)) --q;
  return q;
}
__device__ __forceinline__ int fmodi(int a, int b) { return a - fdivi(a, b) * b; }

__global__ void k_coords(const int* __restrict__ unq, int* __restrict__ out) {
  int i = blockIdx.x * blockDim.x + threadIdx.x;
  if (i >= NPTS) return;
  int u = unq[i];                                    // -1 fill handled by floor semantics
  int b = fdivi(u, SCALE_XYZ);
  int x = fdivi(fmodi(u, SCALE_XYZ), SCALE_YZ);
  int y = fdivi(fmodi(u, SCALE_YZ), SCALE_Z);
  int z = fmodi(u, SCALE_Z);
  out[i*4+0] = b; out[i*4+1] = z; out[i*4+2] = y; out[i*4+3] = x;  // [b,z,y,x]
}

// ================================================================ launch ====
extern "C" void kernel_launch(void* const* d_in, const int* in_sizes, int n_in,
                              void* d_out, int out_size, void* d_ws, size_t ws_size,
                              hipStream_t stream) {
  (void)in_sizes; (void)n_in; (void)out_size; (void)ws_size;
  const float* pts = (const float*)d_in[0];
  const float* W0  = (const float*)d_in[1];
  const float* g0  = (const float*)d_in[2];
  const float* b0  = (const float*)d_in[3];
  const float* W1  = (const float*)d_in[4];
  const float* g1  = (const float*)d_in[5];
  const float* b1  = (const float*)d_in[6];

  char* ws = (char*)d_ws;
  size_t off = 0;
  auto alloc = [&](size_t bytes) -> char* {
    char* p = ws + off;
    off = (off + bytes + 255) & ~(size_t)255;
    return p;
  };
  unsigned* rank    = (unsigned*)alloc((size_t)NKEYS * 4);      // flags -> excl prefix
  unsigned* tsum    = (unsigned*)alloc((size_t)NTILES * 4);
  int*      key     = (int*)     alloc((size_t)NPTS * 4);
  int*      inv     = (int*)     alloc((size_t)NPTS * 4);
  int*      counts  = (int*)     alloc((size_t)NPTS * 4);
  float*    sums    = (float*)   alloc((size_t)NPTS * 12);
  int*      unq     = (int*)     alloc((size_t)NPTS * 4);
  _Float16* feats10 = (_Float16*)alloc((size_t)NPTS * 32 * 2);  // K padded 10->32
  _Float16* x0      = (_Float16*)alloc((size_t)NPTS * 64 * 2);
  int*      xmax0   = (int*)     alloc((size_t)NPTS * 64 * 4);  // f32 bits, >=0
  _Float16* x1      = (_Float16*)alloc((size_t)NPTS * 128 * 2);
  _Float16* W0p     = (_Float16*)alloc((size_t)4 * 512 * 2);
  _Float16* W1p     = (_Float16*)alloc((size_t)32 * 512 * 2);
  float*    st      = (float*)   alloc((size_t)768 * 4);
  float *chsum0 = st,       *chsq0 = st+64,  *scale0 = st+128, *shift0 = st+192;
  float *chsum1 = st+256,   *chsq1 = st+384, *scale1 = st+512, *shift1 = st+640;

  float* outF = (float*)d_out;                   // N x 128 features
  int*   outC = (int*)d_out + (size_t)NPTS*128;  // N x 4 voxel coords (int bits)

  const int T = 256;
  // ---- zero-init (grid-stride fills; graph-capture safe) ----
  k_fill_u4<<<2048, T, 0, stream>>>((uint4*)rank,   0u,          NKEYS/4);
  k_fill_u4<<<256,  T, 0, stream>>>((uint4*)counts, 0u,          NPTS/4);
  k_fill_u4<<<256,  T, 0, stream>>>((uint4*)sums,   0u,          (long)NPTS*3/4);
  k_fill_u4<<<256,  T, 0, stream>>>((uint4*)unq,    0xFFFFFFFFu, NPTS/4);     // -1 fill
  k_fill_u4<<<2048, T, 0, stream>>>((uint4*)xmax0,  0u,          (long)NPTS*64/4);
  k_fill_u4<<<1,    T, 0, stream>>>((uint4*)st,     0u,          768/4);
  k_fill_u4<<<2048, T, 0, stream>>>((uint4*)outF,   0u,          (long)NPTS*128/4);

  // ---- voxel keys, dense rank scan (L2-resident), inverse ----
  const int gN = (NPTS + T - 1) / T;
  k_keys <<<gN,     T, 0, stream>>>(pts, key, rank);
  k_scanA<<<NTILES, T, 0, stream>>>(rank, tsum);
  k_scanB<<<1,      T, 0, stream>>>(tsum, NTILES);
  k_scanC<<<NTILES, T, 0, stream>>>(rank, tsum);
  k_inv  <<<gN,     T, 0, stream>>>(key, rank, inv, counts, sums, unq, pts);
  k_feats<<<gN,     T, 0, stream>>>(pts, inv, counts, sums, feats10);

  // ---- weight fragments ----
  k_prepw0<<<8,  T, 0, stream>>>(W0, W0p);
  k_prepw1<<<64, T, 0, stream>>>(W1, W1p);

  // ---- PFN layer 0: stats pass, BN coeffs, apply pass ----
  const int gG = NPTS / 64;                      // 4 waves/block, 16 rows/wave
  k_gemm0_stats<<<gG, 128, 0, stream>>>(feats10, W0p, chsum0, chsq0);
  k_stats<<<1, 128, 0, stream>>>(chsum0, chsq0, g0, b0, scale0, shift0, 64);
  k_gemm0_apply<<<gG, 128, 0, stream>>>(feats10, W0p, scale0, shift0, inv, x0, xmax0);
  k_concat<<<(int)((long)NPTS*16/T), T, 0, stream>>>(x0, (const float*)xmax0, inv, x1);

  // ---- PFN layer 1: stats pass, BN coeffs, apply+segment-max into output ----
  k_gemm1_stats<<<gG, 128, 0, stream>>>(x1, W1p, chsum1, chsq1);
  k_stats<<<1, 128, 0, stream>>>(chsum1, chsq1, g1, b1, scale1, shift1, 128);
  k_gemm1_apply<<<gG, 128, 0, stream>>>(x1, W1p, scale1, shift1, inv, outF);

  // ---- voxel coords [b,z,y,x] ----
  k_coords<<<gN, T, 0, stream>>>(unq, outC);
}
